// MultiHeadAttention_2259152798076
// MI455X (gfx1250) — matrix operand order repaired and checked by reading, not compiled, not run on hardware
//
#include <hip/hip_runtime.h>

typedef __attribute__((ext_vector_type(16))) _Float16 v16h;
typedef __attribute__((ext_vector_type(8)))  float    v8f;

#define BATCH  8
#define SEQ    2048
#define DMODEL 128
#define NHEAD  8
#define DK     16

// p = 2^(score * 0.25 * log2(e) - PSHIFT); constant shift cancels in the
// softmax normalization, keeps exp results comfortably inside f16 range.
#define QLOG2E 0.360673760222240851f   // 0.25 * log2(e)
#define PSHIFT 12.0f

// LDS probability tile: 16 rows x 32 cols, row stride padded to 36 halves
// (18 words) so the two lane-halves hit disjoint bank sets (64-bank LDS).
#define LDSW 36

__device__ __forceinline__ v8f wmma_f16(v16h a, v16h b, v8f c) {
  // D = A(16x32 f16) * B(32x16 f16) + C(16x16 f32)
  return __builtin_amdgcn_wmma_f32_16x16x32_f16(
      /*neg_a=*/false, a, /*neg_b=*/false, b,
      /*c_mod=*/(short)0, c, /*reuse_a=*/false, /*reuse_b=*/false);
}

// Raw v_exp_f32: our arguments are ~[-30, +7], far above the 2^-126 denormal
// cliff that the libm exp2f fixup (cmp/cndmask chains) protects against.
__device__ __forceinline__ float fast_exp2(float x) {
  return __builtin_amdgcn_exp2f(x);
}

// ---------------------------------------------------------------------------
// GEMM: Out[16384 x 128] = A[16384 x 128] @ W^T, W [128 x 128] row-major.
// One wave per 16x16 output tile, 4 chained WMMAs over K=128.
// A_IS_F16: A is _Float16 (ctx scratch) else float (x input).
// OUT_F32 : write float (final output) else _Float16 (kqv scratch).
// WRITE_T : additionally write a d-major copy OutT[b][head][d][s] so the
//           attention kernel can load V columns with wide contiguous loads.
// ---------------------------------------------------------------------------
template <bool A_IS_F16, bool OUT_F32, bool WRITE_T>
__global__ __launch_bounds__(32)
void gemm128_nt_kernel(const void* __restrict__ Aptr,
                       const float* __restrict__ W,
                       void* __restrict__ Out,
                       _Float16* __restrict__ OutT) {
  const int m0 = blockIdx.x * 16;
  const int n0 = blockIdx.y * 16;
  const int lane = threadIdx.x & 31;
  const int hi = lane >> 4;     // lane half
  const int ln = lane & 15;

  v8f acc = {};
#pragma unroll
  for (int k0 = 0; k0 < DMODEL; k0 += 32) {
    // ---- A operand: 16x32 f16, row M = m0+ln, K = 16*hi + 2v + h
    v16h a;
#pragma unroll
    for (int v = 0; v < 8; ++v) {
      const int kk = 16 * hi + 2 * v;
      if (A_IS_F16) {
        const _Float16* ar = (const _Float16*)Aptr + (size_t)(m0 + ln) * DMODEL + k0;
        a[2 * v]     = ar[kk];
        a[2 * v + 1] = ar[kk + 1];
      } else {
        const float* ar = (const float*)Aptr + (size_t)(m0 + ln) * DMODEL + k0;
        a[2 * v]     = (_Float16)ar[kk];
        a[2 * v + 1] = (_Float16)ar[kk + 1];
      }
    }
    // ---- B operand: 32x16 f16, col N = n0+ln, K = 16*hi + 2v + h
    // B[k][n] = W[n0+n][k0+k]  (row of W contiguous in k -> b128 loads)
    const float* wr = W + (size_t)(n0 + ln) * DMODEL + k0 + 16 * hi;
    v16h b;
#pragma unroll
    for (int v = 0; v < 8; ++v) {
      b[2 * v]     = (_Float16)wr[2 * v];
      b[2 * v + 1] = (_Float16)wr[2 * v + 1];
    }
    acc = wmma_f16(a, b, acc);
  }

  // ---- store: C/D layout: reg r, lane -> (M = r + 8*hi, N = ln)
#pragma unroll
  for (int r = 0; r < 8; ++r) {
    const int row = r + 8 * hi;
    const int g = m0 + row;                       // global row in [0, B*S)
    const size_t off = (size_t)g * DMODEL + n0 + ln;
    const float val = acc[r];
    if (OUT_F32)
      ((float*)Out)[off] = val;
    else
      ((_Float16*)Out)[off] = (_Float16)val;
    if (WRITE_T) {
      const int bb = g >> 11;                     // g / SEQ
      const int s  = g & (SEQ - 1);               // g % SEQ
      const int head = n0 >> 4;
      OutT[(((size_t)bb * NHEAD + head) * DK + ln) * SEQ + s] = (_Float16)val;
    }
  }
}

// ---------------------------------------------------------------------------
// Flash attention, one wave per (b, head, 32-query block = 2 x 16-row tiles).
// kqv  f16 [B, S, 128]   : Q rows and K rows (row-major, wide loads)
// kqvT f16 [B, H, DK, S] : V columns (key-major, wide contiguous loads)
// Fixed-shift softmax (no running max), row-sums via a WMMA against an
// all-ones B matrix -> zero cross-lane shuffles in the hot loop.
//
// Key-interleave: inside each 32-key tile, P columns / V rows use the
// permuted order  c  <->  key (c&1)*16 + (c>>1).  P@V and rowsum(P) are
// invariant under any consistent key permutation; this makes each lane's
// (p0, p1) pair LDS-adjacent -> single packed ds_store_b32.
//
// Per 32-key step: K^T/V fetched once, reused by 2 query tiles:
//   2 x (2 QK^T WMMA + 1 P@V WMMA + 1 P@1 WMMA) = 8 WMMAs per fetch.
// ---------------------------------------------------------------------------
__global__ __launch_bounds__(32)
void flash_attn_kernel(const _Float16* __restrict__ kqv,
                       const _Float16* __restrict__ kqvT,
                       _Float16* __restrict__ ctx) {
  const int q0   = blockIdx.x * 32;     // first query row of this wave
  const int head = blockIdx.y;
  const int b    = blockIdx.z;
  const int lane = threadIdx.x & 31;
  const int hi = lane >> 4;
  const int ln = lane & 15;

  const _Float16* base   = kqv  + (size_t)b * SEQ * DMODEL + head * DK;
  const _Float16* vbaseT = kqvT + (((size_t)b * NHEAD + head) * DK + ln) * SEQ;

  __shared__ _Float16 ldsP[16 * LDSW];  // padded probability tile (1152 B)

  // ---- ones B operand (32x16, every element 1.0) for row-sum WMMA
  v16h ones;
#pragma unroll
  for (int i = 0; i < 16; ++i) ones[i] = (_Float16)1.0f;

  // ---- Q tiles as A operands: 16 x 32, K = d (0..15 real, 16..31 zero-pad)
  v16h qa[2];
#pragma unroll
  for (int t = 0; t < 2; ++t) {
    const _Float16* qr = base + (size_t)(q0 + t * 16 + ln) * DMODEL;
#pragma unroll
    for (int v = 0; v < 8; ++v) {
      if (hi == 0) {
        const int kb = 2 * v;                   // d index, < 16
        qa[t][2 * v]     = qr[kb];
        qa[t][2 * v + 1] = qr[kb + 1];
      } else {
        qa[t][2 * v]     = (_Float16)0.0f;      // pad d = 16..31
        qa[t][2 * v + 1] = (_Float16)0.0f;
      }
    }
  }

  v8f acc[2]  = {{}, {}};   // running P@V per query tile, C/D layout
  v8f lsum[2] = {{}, {}};   // running row-sums of P, replicated across columns

  for (int kt = 0; kt < SEQ / 32; ++kt) {
    if (kt + 1 < SEQ / 32) {
      __builtin_prefetch(base + (size_t)((kt + 1) * 32) * DMODEL, 0, 1);
      __builtin_prefetch(vbaseT + (kt + 1) * 32, 0, 1);
    }

    // ---- K^T operand tiles (32x16 B layout): rows = d (pad >= 16), cols = key
    v16h bk0 = {}, bk1 = {};
    if (hi == 0) {                                // rows d = 0..15 only
      const _Float16* kr0 = base + (size_t)(kt * 32 + ln) * DMODEL;
      const _Float16* kr1 = base + (size_t)(kt * 32 + 16 + ln) * DMODEL;
#pragma unroll
      for (int v = 0; v < 8; ++v) {
        bk0[2 * v]     = kr0[2 * v];
        bk0[2 * v + 1] = kr0[2 * v + 1];
        bk1[2 * v]     = kr1[2 * v];
        bk1[2 * v + 1] = kr1[2 * v + 1];
      }
    }

    // ---- V tile as B operand (32x16) in interleaved key order:
    // row kk -> key 16*(kk&1) + 8*hi + (kk>>1)&7.  Even halves come from
    // keys [8hi, 8hi+8), odd halves from [16+8hi, 16+8hi+8): two contiguous
    // 16-byte runs per lane (b128 loads) + half-word interleave.
    v16h bv;
    {
      const _Float16* vlo = vbaseT + kt * 32 + 8 * hi;        // kk even
      const _Float16* vhio = vbaseT + kt * 32 + 16 + 8 * hi;  // kk odd
#pragma unroll
      for (int v = 0; v < 8; ++v) {
        bv[2 * v]     = vlo[v];
        bv[2 * v + 1] = vhio[v];
      }
    }

    // ---- per query tile: scores -> exp -> LDS layout swap -> accumulate
#pragma unroll
    for (int t = 0; t < 2; ++t) {
      v8f s0 = {}, s1 = {};
      s0 = wmma_f16(qa[t], bk0, s0);              // scores, keys kt*32 + 0..15
      s1 = wmma_f16(qa[t], bk1, s1);              // scores, keys kt*32 + 16..31

      // fixed-shift softmax numerator: p = 2^(score*0.25*log2e - PSHIFT)
      // s0 col ln -> key ln -> interleaved col 2*ln; s1 -> col 2*ln+1.
#pragma unroll
      for (int r = 0; r < 8; ++r) {
        const float p0 = fast_exp2(fmaf(s0[r], QLOG2E, -PSHIFT));
        const float p1 = fast_exp2(fmaf(s1[r], QLOG2E, -PSHIFT));
        const int row = r + 8 * hi;               // C/D layout row
        ldsP[row * LDSW + 2 * ln]     = (_Float16)p0;   // packed b32 store
        ldsP[row * LDSW + 2 * ln + 1] = (_Float16)p1;
      }
      __syncthreads();

      // reload P in A layout: M = ln, K(col) = 0..31 (interleaved key order)
      v16h pa;
      {
        const _Float16* pr = ldsP + ln * LDSW;
#pragma unroll
        for (int v = 0; v < 8; ++v) {
          const int kb = 16 * hi + 2 * v;
          pa[2 * v]     = pr[kb];
          pa[2 * v + 1] = pr[kb + 1];
        }
      }
      __syncthreads();

      acc[t]  = wmma_f16(pa, bv,   acc[t]);       // ctx  += P @ V
      lsum[t] = wmma_f16(pa, ones, lsum[t]);      // rowsum(P) in every column
    }
  }

  // ---- normalize (2^-PSHIFT cancels) and store ctx (f16)
#pragma unroll
  for (int t = 0; t < 2; ++t) {
#pragma unroll
    for (int r = 0; r < 8; ++r) {
      const float inv = 1.0f / lsum[t][r];
      const int row = r + 8 * hi;
      ctx[((size_t)b * SEQ + q0 + t * 16 + row) * DMODEL + head * DK + ln] =
          (_Float16)(acc[t][r] * inv);
    }
  }
}

extern "C" void kernel_launch(void* const* d_in, const int* in_sizes, int n_in,
                              void* d_out, int out_size, void* d_ws, size_t ws_size,
                              hipStream_t stream) {
  (void)in_sizes; (void)n_in; (void)out_size; (void)ws_size;
  const float* x   = (const float*)d_in[0];
  const float* W_k = (const float*)d_in[1];
  // d_in[2] = W_q, d_in[3] = W_v : unused (faithful to reference source bug)
  const float* W_o = (const float*)d_in[4];
  float* out = (float*)d_out;

  const size_t elems = (size_t)BATCH * SEQ * DMODEL;          // 2 Mi elements
  _Float16* kqv  = (_Float16*)d_ws;                           // 4 MB
  _Float16* kqvT = kqv + elems;                               // 4 MB (d-major)
  _Float16* ctx  = kqvT + elems;                              // 4 MB

  dim3 gProj(BATCH * SEQ / 16, DMODEL / 16, 1);   // 1024 x 8 tiles
  dim3 gAttn(SEQ / 32, NHEAD, BATCH);             // 64 x 8 x 8 waves

  // 1) kqv = x @ W_k^T  (f32 in, f16 out, plus d-major transposed copy)
  gemm128_nt_kernel<false, false, true><<<gProj, 32, 0, stream>>>(x, W_k, kqv, kqvT);
  // 2) flash attention per (b, head, 32-query block)
  flash_attn_kernel<<<gAttn, 32, 0, stream>>>(kqv, kqvT, ctx);
  // 3) out = ctx @ W_o^T  (f16 in, f32 out)
  gemm128_nt_kernel<true, true, false><<<gProj, 32, 0, stream>>>(ctx, W_o, out, nullptr);
}
